// Attention_62294205661952
// MI455X (gfx1250) — matrix operand order repaired and checked by reading, not compiled, not run on hardware
//
#include <hip/hip_runtime.h>

typedef __attribute__((ext_vector_type(16))) __bf16 v16bf;
typedef __attribute__((ext_vector_type(8)))  float  v8f;
typedef unsigned int   u32;
typedef unsigned short u16;

union Frag { v16bf v; u32 u[8]; };

#define BATCH  4
#define SEQ    512
#define DMODEL 4096
#define NH     32
#define NKV    8
#define HD     128
#define STARTP 512
#define TTOT   1024
#define NREP   (NH / NKV)

__device__ __forceinline__ u16 f2bf(float f) {
  union { float f; u32 u; } x; x.f = f;
  u32 u = x.u;
  u32 r = (u + 0x7FFFu + ((u >> 16) & 1u)) >> 16;  // round-to-nearest-even
  return (u16)r;
}
__device__ __forceinline__ u32 pack2(float a, float b) {
  return (u32)f2bf(a) | ((u32)f2bf(b) << 16);
}

// ---------------------------------------------------------------------------
// fp32 -> bf16 bulk convert (vectorized 4 elems/thread)
// ---------------------------------------------------------------------------
__global__ void cvt_bf16_kernel(const float* __restrict__ src,
                                u16* __restrict__ dst, int n4) {
  const int i = blockIdx.x * blockDim.x + threadIdx.x;
  if (i >= n4) return;
  const float4 v = ((const float4*)src)[i];
  uint2 o;
  o.x = pack2(v.x, v.y);
  o.y = pack2(v.z, v.w);
  ((uint2*)dst)[i] = o;
}

// ---------------------------------------------------------------------------
// C[M,N] = A[M,K] * W[N,K]^T   (A, W bf16 row-major; f32 accum)
// 128 threads (4 waves); tile 128x64, BK=32; LDS double-buffered;
// next tile's global loads issued before current tile's WMMAs.
// ---------------------------------------------------------------------------
template <int N, int K>
__global__ __launch_bounds__(128) void gemm_bf16(const u16* __restrict__ A,
                                                 const u16* __restrict__ W,
                                                 float* __restrict__ C) {
  __shared__ u32 lA[2][128][16];  // 2 x (128 rows x 32 bf16) = 16 KiB
  __shared__ u32 lB[2][64][16];   // 2 x ( 64 rows x 32 bf16) =  8 KiB
  const int tid  = threadIdx.x;
  const int wave = tid >> 5;
  const int lane = tid & 31;
  const int ln   = lane & 15;
  const int hi   = lane >> 4;
  const int m0 = blockIdx.y * 128;
  const int n0 = blockIdx.x * 64;
  constexpr int Ku = K / 2;  // row stride in u32

  const u32* Au = (const u32*)A;
  const u32* Wu = (const u32*)W;

  // per-thread staging coordinates: chunk it covers row rowT+32*it, cols 8*cT..
  const int rowT = tid >> 2;
  const int cT   = tid & 3;
  const u32* gA[4];
  #pragma unroll
  for (int it = 0; it < 4; ++it)
    gA[it] = Au + (size_t)(m0 + rowT + 32 * it) * Ku + 4 * cT;
  const u32* gB[2];
  #pragma unroll
  for (int it = 0; it < 2; ++it)
    gB[it] = Wu + (size_t)(n0 + rowT + 32 * it) * Ku + 4 * cT;

  v8f zero = {};
  v8f acc[2][4];
  #pragma unroll
  for (int g = 0; g < 2; ++g)
    #pragma unroll
    for (int j = 0; j < 4; ++j) acc[g][j] = zero;

  // prologue: stage tile 0 into buffer 0
  uint4 rA[4], rB[2];
  #pragma unroll
  for (int it = 0; it < 4; ++it) rA[it] = *(const uint4*)(gA[it]);
  #pragma unroll
  for (int it = 0; it < 2; ++it) rB[it] = *(const uint4*)(gB[it]);
  #pragma unroll
  for (int it = 0; it < 4; ++it) *(uint4*)&lA[0][rowT + 32 * it][4 * cT] = rA[it];
  #pragma unroll
  for (int it = 0; it < 2; ++it) *(uint4*)&lB[0][rowT + 32 * it][4 * cT] = rB[it];
  __syncthreads();

  int cur = 0;
  for (int k0 = 0; k0 < K; k0 += 32, cur ^= 1) {
    const bool hasNext = (k0 + 32) < K;
    const int kuN = (k0 + 32) >> 1;
    if (hasNext) {
      // issue next tile's global loads early; wait lands after the WMMAs
      #pragma unroll
      for (int it = 0; it < 4; ++it) rA[it] = *(const uint4*)(gA[it] + kuN);
      #pragma unroll
      for (int it = 0; it < 2; ++it) rB[it] = *(const uint4*)(gB[it] + kuN);
      __builtin_prefetch(gA[0] + kuN + 16, 0, 3);  // warm next-next tile
      __builtin_prefetch(gB[0] + kuN + 16, 0, 3);
    }

    // compute from current buffer
    Frag a0, a1;
    {
      const u32* s0 = &lA[cur][wave * 32 + ln][0];
      const u32* s1 = &lA[cur][wave * 32 + 16 + ln][0];
      #pragma unroll
      for (int i = 0; i < 8; ++i) {
        const int off = i + 8 * hi;  // A contiguous K
        a0.u[i] = s0[off];
        a1.u[i] = s1[off];
      }
    }
    #pragma unroll
    for (int j = 0; j < 4; ++j) {
      Frag bf;
      const u32* src = &lB[cur][j * 16 + ln][0];
      #pragma unroll
      for (int i = 0; i < 8; ++i) bf.u[i] = src[i + 8 * hi];  // B contiguous K
      acc[0][j] = __builtin_amdgcn_wmma_f32_16x16x32_bf16(
          false, a0.v, false, bf.v, (short)0, acc[0][j], false, false);
      acc[1][j] = __builtin_amdgcn_wmma_f32_16x16x32_bf16(
          false, a1.v, false, bf.v, (short)0, acc[1][j], false, false);
    }

    // stage next tile into the other buffer (no conflict with current reads)
    if (hasNext) {
      const int nxt = cur ^ 1;
      #pragma unroll
      for (int it = 0; it < 4; ++it) *(uint4*)&lA[nxt][rowT + 32 * it][4 * cT] = rA[it];
      #pragma unroll
      for (int it = 0; it < 2; ++it) *(uint4*)&lB[nxt][rowT + 32 * it][4 * cT] = rB[it];
    }
    __syncthreads();
  }

  #pragma unroll
  for (int g = 0; g < 2; ++g)
    #pragma unroll
    for (int j = 0; j < 4; ++j)
      #pragma unroll
      for (int r = 0; r < 8; ++r) {
        const int m = m0 + wave * 32 + g * 16 + r + 8 * hi;
        const int n = n0 + j * 16 + ln;
        C[(size_t)m * N + n] = acc[g][j][r];
      }
}

// ---------------------------------------------------------------------------
// RoPE on Q -> bf16 (B,S,H,HD); 1/sqrt(HD) folded in
// ---------------------------------------------------------------------------
__global__ void rope_q_kernel(const float* __restrict__ xq,
                              const float* __restrict__ cosb,
                              const float* __restrict__ sinb,
                              u16* __restrict__ qb) {
  size_t idx = (size_t)blockIdx.x * blockDim.x + threadIdx.x;
  const size_t total = (size_t)BATCH * SEQ * NH * (HD / 2);
  if (idx >= total) return;
  const int d2 = idx & 63;
  size_t r = idx >> 6;
  const int h = r % NH;  r /= NH;
  const int s = r % SEQ;
  const int bi = r / SEQ;
  const float scale = 0.08838834764831845f;  // 1/sqrt(128)
  const size_t base = ((size_t)((bi * SEQ + s) * NH + h)) * HD + 2 * d2;
  const float a = xq[base], b2 = xq[base + 1];
  const float c = cosb[s * 64 + d2], si = sinb[s * 64 + d2];
  qb[base]     = f2bf((a * c - b2 * si) * scale);
  qb[base + 1] = f2bf((a * si + b2 * c) * scale);
}

// ---------------------------------------------------------------------------
// K buffer: cache (raw) ++ rotated new keys -> bf16 (B,T,HKV,HD)
// ---------------------------------------------------------------------------
__global__ void k_build_kernel(const float* __restrict__ xk,
                               const float* __restrict__ ck,
                               const float* __restrict__ cosb,
                               const float* __restrict__ sinb,
                               u16* __restrict__ kb) {
  size_t idx = (size_t)blockIdx.x * blockDim.x + threadIdx.x;
  const size_t total = (size_t)BATCH * TTOT * NKV * (HD / 2);
  if (idx >= total) return;
  const int d2 = idx & 63;
  size_t r = idx >> 6;
  const int h = r % NKV;  r /= NKV;
  const int t = r % TTOT;
  const int bi = r / TTOT;
  const size_t obase = ((size_t)((bi * TTOT + t) * NKV + h)) * HD + 2 * d2;
  if (t < STARTP) {
    const size_t cbase = ((size_t)((bi * STARTP + t) * NKV + h)) * HD + 2 * d2;
    kb[obase]     = f2bf(ck[cbase]);
    kb[obase + 1] = f2bf(ck[cbase + 1]);
  } else {
    const int s = t - STARTP;
    const size_t xbase = ((size_t)((bi * SEQ + s) * NKV + h)) * HD + 2 * d2;
    const float a = xk[xbase], b2 = xk[xbase + 1];
    const float c = cosb[s * 64 + d2], si = sinb[s * 64 + d2];
    kb[obase]     = f2bf(a * c - b2 * si);
    kb[obase + 1] = f2bf(a * si + b2 * c);
  }
}

// ---------------------------------------------------------------------------
// V transposed buffer: bf16 (B,HKV,HD,T)   (key-fastest for B-fragments)
// ---------------------------------------------------------------------------
__global__ void v_build_kernel(const float* __restrict__ xv,
                               const float* __restrict__ cv,
                               u16* __restrict__ vt) {
  size_t idx = (size_t)blockIdx.x * blockDim.x + threadIdx.x;
  const size_t total = (size_t)BATCH * NKV * HD * TTOT;
  if (idx >= total) return;
  const int t = idx % TTOT;
  size_t r = idx / TTOT;
  const int d = r % HD;  r /= HD;
  const int h = r % NKV;
  const int bi = r / NKV;
  float val;
  if (t < STARTP)
    val = cv[((size_t)((bi * STARTP + t) * NKV + h)) * HD + d];
  else
    val = xv[((size_t)((bi * SEQ + (t - STARTP)) * NKV + h)) * HD + d];
  vt[((size_t)((bi * NKV + h) * HD + d)) * TTOT + t] = f2bf(val);
}

// ---------------------------------------------------------------------------
// Causal flash attention: one wave per (b, h, 16-query tile); bf16 output
// ---------------------------------------------------------------------------
__global__ __launch_bounds__(32) void flash_fwd(const u16* __restrict__ Qb,
                                                const u16* __restrict__ Kb,
                                                const u16* __restrict__ Vt,
                                                u16* __restrict__ Ob) {
  __shared__ u32 lPs[256];              // 16 rows x 32 keys of bf16 probs
  u16* lP = (u16*)lPs;
  const u32* lpu = lPs;

  const int lane = threadIdx.x;
  const int ln = lane & 15, hi = lane >> 4;
  const int bi = blockIdx.z, h = blockIdx.y;
  const int q0 = blockIdx.x * 16;
  const int hk = h / NREP;

  const u32* qbu = (const u32*)Qb;
  const u32* kbu = (const u32*)Kb;
  const u32* vtu = (const u32*)Vt;

  // Q fragments (A-layout): row q0+ln, dim chunks of 32
  Frag qf[4];
  {
    const u32* qrow = qbu + ((size_t)((bi * SEQ + q0 + ln) * NH + h)) * (HD / 2);
    #pragma unroll
    for (int f = 0; f < 4; ++f)
      #pragma unroll
      for (int i = 0; i < 8; ++i)
        qf[f].u[i] = qrow[16 * f + i + 8 * hi];
  }

  v8f zero = {};
  v8f o[8];
  #pragma unroll
  for (int d = 0; d < 8; ++d) o[d] = zero;
  float mrow[8], lrow[8];
  #pragma unroll
  for (int r = 0; r < 8; ++r) { mrow[r] = -INFINITY; lrow[r] = 0.f; }

  const int npairs = (STARTP + q0 + 15) / 32 + 1;

  for (int j = 0; j < npairs; ++j) {
    const int kb0 = 32 * j;
    // --- scores: two 16x16 tiles over 32 keys (scale pre-folded into Q) ---
    v8f s0 = zero, s1 = zero;
    {
      const u32* kr0 = kbu + ((size_t)((bi * TTOT + kb0 + ln) * NKV + hk)) * (HD / 2);
      const u32* kr1 = kbu + ((size_t)((bi * TTOT + kb0 + 16 + ln) * NKV + hk)) * (HD / 2);
      #pragma unroll
      for (int f = 0; f < 4; ++f) {
        Frag kf0, kf1;
        #pragma unroll
        for (int i = 0; i < 8; ++i) {
          kf0.u[i] = kr0[16 * f + i + 8 * hi];
          kf1.u[i] = kr1[16 * f + i + 8 * hi];
        }
        s0 = __builtin_amdgcn_wmma_f32_16x16x32_bf16(false, qf[f].v, false, kf0.v,
                                                     (short)0, s0, false, false);
        s1 = __builtin_amdgcn_wmma_f32_16x16x32_bf16(false, qf[f].v, false, kf1.v,
                                                     (short)0, s1, false, false);
      }
    }
    // --- causal mask + online softmax (rows span 16-lane halves) ---
    float alpha[8];
    #pragma unroll
    for (int r = 0; r < 8; ++r) {
      const int qabs = STARTP + q0 + r + 8 * hi;
      float sc0 = s0[r];
      float sc1 = s1[r];
      if (kb0 + ln > qabs)      sc0 = -INFINITY;
      if (kb0 + 16 + ln > qabs) sc1 = -INFINITY;
      float mx = fmaxf(sc0, sc1);
      #pragma unroll
      for (int off = 1; off < 16; off <<= 1) mx = fmaxf(mx, __shfl_xor(mx, off, 32));
      const float mnew = fmaxf(mrow[r], mx);
      alpha[r] = __expf(mrow[r] - mnew);
      mrow[r] = mnew;
      const float p0 = __expf(sc0 - mnew);
      const float p1 = __expf(sc1 - mnew);
      float rs = p0 + p1;
      #pragma unroll
      for (int off = 1; off < 16; off <<= 1) rs += __shfl_xor(rs, off, 32);
      lrow[r] = lrow[r] * alpha[r] + rs;
      lP[(r + 8 * hi) * 32 + ln]      = f2bf(p0);   // C-layout -> LDS
      lP[(r + 8 * hi) * 32 + 16 + ln] = f2bf(p1);
    }
    __syncthreads();

    // rescale running output
    #pragma unroll
    for (int d = 0; d < 8; ++d)
      #pragma unroll
      for (int r = 0; r < 8; ++r) o[d][r] *= alpha[r];

    // P fragment (A-layout) from LDS
    Frag pf;
    #pragma unroll
    for (int i = 0; i < 8; ++i)
      pf.u[i] = lpu[ln * 16 + i + 8 * hi];

    // --- O += P (16x32) x V (32x128) ---
    #pragma unroll
    for (int d = 0; d < 8; ++d) {
      Frag vf;
      const u32* vr = vtu + ((size_t)((bi * NKV + hk) * HD + 16 * d + ln)) * (TTOT / 2);
      #pragma unroll
      for (int i = 0; i < 8; ++i) vf.u[i] = vr[16 * j + i + 8 * hi];
      o[d] = __builtin_amdgcn_wmma_f32_16x16x32_bf16(false, pf.v, false, vf.v,
                                                     (short)0, o[d], false, false);
    }
    __syncthreads();
  }

  // normalize and store (B,S,H,HD) bf16 (feeds the wo GEMM directly)
  #pragma unroll
  for (int r = 0; r < 8; ++r) {
    const float inv = 1.f / lrow[r];
    const int s = q0 + r + 8 * hi;
    #pragma unroll
    for (int d = 0; d < 8; ++d)
      Ob[(size_t)(((bi * SEQ + s) * NH + h) * HD + 16 * d + ln)] = f2bf(o[d][r] * inv);
  }
}

// ---------------------------------------------------------------------------
extern "C" void kernel_launch(void* const* d_in, const int* in_sizes, int n_in,
                              void* d_out, int out_size, void* d_ws, size_t ws_size,
                              hipStream_t stream) {
  (void)in_sizes; (void)n_in; (void)out_size; (void)ws_size;
  const float* x  = (const float*)d_in[0];
  const float* wq = (const float*)d_in[1];
  const float* wk = (const float*)d_in[2];
  const float* wv = (const float*)d_in[3];
  const float* wo = (const float*)d_in[4];
  const float* fc = (const float*)d_in[5];
  const float* fs = (const float*)d_in[6];
  const float* ck = (const float*)d_in[7];
  const float* cv = (const float*)d_in[8];
  // d_in[9] = start_pos (constant 512 per setup)

  char* ws = (char*)d_ws;
  u16*   xb  = (u16*)  (ws + 0ull);          // 2048*4096   bf16 = 16 MiB
  u16*   wqb = (u16*)  (ws + 16777216ull);   // 4096*4096   bf16 = 32 MiB
  u16*   wkb = (u16*)  (ws + 50331648ull);   // 1024*4096   bf16 =  8 MiB
  u16*   wvb = (u16*)  (ws + 58720256ull);   //                     8 MiB
  u16*   wob = (u16*)  (ws + 67108864ull);   // 4096*4096   bf16 = 32 MiB
  float* xq  = (float*)(ws + 100663296ull);  // 2048*4096   f32  = 32 MiB
  float* xk  = (float*)(ws + 134217728ull);  // 2048*1024   f32  =  8 MiB
  float* xv  = (float*)(ws + 142606336ull);  //                     8 MiB
  u16*   qb  = (u16*)  (ws + 150994944ull);  // 2048*4096   bf16 = 16 MiB
  u16*   kb  = (u16*)  (ws + 167772160ull);  // 4*1024*8*128 bf16 = 8 MiB
  u16*   vt  = (u16*)  (ws + 176160768ull);  //                     8 MiB
  u16*   ao  = (u16*)  (ws + 184549376ull);  // 2048*4096   bf16 = 16 MiB
  float* y   = (float*)d_out;

  const int M = BATCH * SEQ;  // 2048

  // one-time bf16 conversions (halve cache/HBM traffic for all GEMMs)
  cvt_bf16_kernel<<<(M * DMODEL / 4) / 256, 256, 0, stream>>>(x, xb, M * DMODEL / 4);
  cvt_bf16_kernel<<<(DMODEL * DMODEL / 4) / 256, 256, 0, stream>>>(wq, wqb, DMODEL * DMODEL / 4);
  cvt_bf16_kernel<<<(NKV * HD * DMODEL / 4) / 256, 256, 0, stream>>>(wk, wkb, NKV * HD * DMODEL / 4);
  cvt_bf16_kernel<<<(NKV * HD * DMODEL / 4) / 256, 256, 0, stream>>>(wv, wvb, NKV * HD * DMODEL / 4);
  cvt_bf16_kernel<<<(DMODEL * DMODEL / 4) / 256, 256, 0, stream>>>(wo, wob, DMODEL * DMODEL / 4);

  // projections
  gemm_bf16<DMODEL, DMODEL><<<dim3(DMODEL / 64, M / 128), 128, 0, stream>>>(xb, wqb, xq);
  gemm_bf16<NKV * HD, DMODEL><<<dim3((NKV * HD) / 64, M / 128), 128, 0, stream>>>(xb, wkb, xk);
  gemm_bf16<NKV * HD, DMODEL><<<dim3((NKV * HD) / 64, M / 128), 128, 0, stream>>>(xb, wvb, xv);

  rope_q_kernel<<<(BATCH * SEQ * NH * 64) / 256, 256, 0, stream>>>(xq, fc, fs, qb);
  k_build_kernel<<<(BATCH * TTOT * NKV * 64) / 256, 256, 0, stream>>>(xk, ck, fc, fs, kb);
  v_build_kernel<<<(BATCH * NKV * HD * TTOT) / 256, 256, 0, stream>>>(xv, cv, vt);

  flash_fwd<<<dim3(SEQ / 16, NH, BATCH), 32, 0, stream>>>(qb, kb, vt, ao);

  // output projection
  gemm_bf16<DMODEL, DMODEL><<<dim3(DMODEL / 64, M / 128), 128, 0, stream>>>(ao, wob, y);
}
